// GRUModel_19722489823246
// MI455X (gfx1250) — matrix operand order repaired and checked by reading, not compiled, not run on hardware
//
#include <hip/hip_runtime.h>
#include <hip/hip_bf16.h>

typedef __attribute__((ext_vector_type(8)))  _Float16 v8h;
typedef __attribute__((ext_vector_type(16))) _Float16 v16h;
typedef __attribute__((ext_vector_type(8)))  float    v8f;

#define H_DIM   400
#define H_PAD   416
#define B_DIM   512
#define D_IN    40
#define D_PAD   64
#define T_STEPS 200
#define C_OUT   12

// ---------------- quantizers (match jnp.round = rint ties-to-even) -------------
__device__ __forceinline__ float sigq(float x) {
    // clip(round(x * 31*0.4)/31, 0, 1)
    return fminf(fmaxf(rintf(x * 12.4f) * (1.0f / 31.0f), 0.0f), 1.0f);
}
__device__ __forceinline__ float tanhq(float x) {
    return fminf(fmaxf(rintf(x * 31.0f) * (1.0f / 31.0f), -1.0f), 1.0f);
}

// ---------------- WMMA helpers --------------------------------------------------
__device__ __forceinline__ v8f wmma_f16(v16h a, v16h b, v8f c) {
    return __builtin_amdgcn_wmma_f32_16x16x32_f16(
        /*neg_a=*/false, a, /*neg_b=*/false, b,
        /*c_mod=*/(short)0, c, /*reuse_a=*/false, /*reuse_b=*/false);
}

// A matrix 16x32 f16 layout (ISA 7.12.2): lane L -> row M=L&15;
// lanes<16: halves 0..7 = K 0..7, halves 8..15 = K 16..23 ; lanes>=16: +8.
__device__ __forceinline__ v16h load_a(const _Float16* __restrict__ base,
                                       int stride, int row, int kb, int lane) {
    const _Float16* p = base + (size_t)row * stride + kb + ((lane >> 4) << 3);
    v8h lo = *(const v8h*)p;          // K = kb+off .. +7
    v8h hi = *(const v8h*)(p + 16);   // K = kb+off+16 .. +23
    return __builtin_shufflevector(lo, hi, 0,1,2,3,4,5,6,7,8,9,10,11,12,13,14,15);
}

// B matrix 32x16 f16: lane L -> col N=L&15; same K order as A:
// lanes<16: K kb..kb+7, kb+16..kb+23 ; lanes>=16: +8. U stored [N][Kpad].
__device__ __forceinline__ v16h load_b(const _Float16* __restrict__ base,
                                       int stride, int col, int kb, int lane) {
    const _Float16* p = base + (size_t)col * stride + kb + ((lane >> 4) << 3);
    v8h lo = *(const v8h*)p;
    v8h hi = *(const v8h*)(p + 16);
    return __builtin_shufflevector(lo, hi, 0,1,2,3,4,5,6,7,8,9,10,11,12,13,14,15);
}

// ---------------- setup kernels -------------------------------------------------
__global__ void cvt_pad_f16(const float* __restrict__ src, _Float16* __restrict__ dst,
                            int rows, int K, int Kpad) {
    int i = blockIdx.x * blockDim.x + threadIdx.x;
    if (i >= rows * Kpad) return;
    int r = i / Kpad, k = i - r * Kpad;
    dst[i] = (k < K) ? (_Float16)src[(size_t)r * K + k] : (_Float16)0.0f;
}
__global__ void zero_f32(float* p, int n) {
    int i = blockIdx.x * blockDim.x + threadIdx.x;
    if (i < n) p[i] = 0.0f;
}
__global__ void zero_f16(_Float16* p, int n) {
    int i = blockIdx.x * blockDim.x + threadIdx.x;
    if (i < n) p[i] = (_Float16)0.0f;
}

// ---------------- GRU step kernel 1: z & r gates --------------------------------
// z = sigq(Xin@Wz^T + h@Uz^T + bz) ; r likewise ; P = r (.) h   (f16, for Uh GEMM)
__global__ __launch_bounds__(256) void gru_gates(
    const _Float16* __restrict__ Xin, int kx,                 // [B, kx] f16 (padded)
    const _Float16* __restrict__ Wz, const _Float16* __restrict__ Wr,  // [H, kx]
    const float* __restrict__ bz, const float* __restrict__ br,
    const _Float16* __restrict__ Hf16,                        // [B, H_PAD]
    const _Float16* __restrict__ Uz, const _Float16* __restrict__ Ur,  // [H, H_PAD]
    const float* __restrict__ Hf32,                           // [B, H]
    float* __restrict__ Zout,                                 // [B, H]
    _Float16* __restrict__ Pout)                              // [B, H_PAD]
{
    const int lane = threadIdx.x & 31;
    const int wave = threadIdx.x >> 5;
    const int tile = blockIdx.x * 8 + wave;        // 800 tiles = 32(M) x 25(N)
    const int tM = (tile & 31) << 4;
    const int tN = (tile >> 5) << 4;
    const int rowA = tM + (lane & 15);
    const int colB = tN + (lane & 15);

    v8f accZ = {}; v8f accR = {};
    for (int kb = 0; kb < kx; kb += 32) {
        v16h a = load_a(Xin, kx, rowA, kb, lane);
        accZ = wmma_f16(a, load_b(Wz, kx, colB, kb, lane), accZ);
        accR = wmma_f16(a, load_b(Wr, kx, colB, kb, lane), accR);
    }
    for (int kb = 0; kb < H_PAD; kb += 32) {
        v16h a = load_a(Hf16, H_PAD, rowA, kb, lane);
        accZ = wmma_f16(a, load_b(Uz, H_PAD, colB, kb, lane), accZ);
        accR = wmma_f16(a, load_b(Ur, H_PAD, colB, kb, lane), accR);
    }

    const int n = colB;
    const float bzv = bz[n], brv = br[n];
    const int rbase = tM + ((lane >> 4) << 3);     // C layout: M = i + 8*(lane>=16)
#pragma unroll
    for (int i = 0; i < 8; ++i) {
        const int row = rbase + i;
        float z = sigq(accZ[i] + bzv);
        float r = sigq(accR[i] + brv);
        Zout[(size_t)row * H_DIM + n] = z;
        Pout[(size_t)row * H_PAD + n] = (_Float16)(r * Hf32[(size_t)row * H_DIM + n]);
    }
}

// ---------------- GRU step kernel 2: candidate + h update -----------------------
// hc = tanhq(Xin@Wh^T + P@Uh^T + bh) ; h = z*h + (1-z)*hc  (writes f32 + f16)
__global__ __launch_bounds__(256) void gru_update(
    const _Float16* __restrict__ Xin, int kx,
    const _Float16* __restrict__ Wh, const float* __restrict__ bh,
    const _Float16* __restrict__ Pf16,                        // [B, H_PAD]
    const _Float16* __restrict__ Uh,                          // [H, H_PAD]
    const float* __restrict__ Zin,                            // [B, H]
    float* __restrict__ Hf32,                                 // [B, H] in/out
    _Float16* __restrict__ Hf16out)                           // [B, H_PAD]
{
    const int lane = threadIdx.x & 31;
    const int wave = threadIdx.x >> 5;
    const int tile = blockIdx.x * 8 + wave;
    const int tM = (tile & 31) << 4;
    const int tN = (tile >> 5) << 4;
    const int rowA = tM + (lane & 15);
    const int colB = tN + (lane & 15);

    v8f acc = {};
    for (int kb = 0; kb < kx; kb += 32) {
        v16h a = load_a(Xin, kx, rowA, kb, lane);
        acc = wmma_f16(a, load_b(Wh, kx, colB, kb, lane), acc);
    }
    for (int kb = 0; kb < H_PAD; kb += 32) {
        v16h a = load_a(Pf16, H_PAD, rowA, kb, lane);
        acc = wmma_f16(a, load_b(Uh, H_PAD, colB, kb, lane), acc);
    }

    const int n = colB;
    const float bhv = bh[n];
    const int rbase = tM + ((lane >> 4) << 3);
#pragma unroll
    for (int i = 0; i < 8; ++i) {
        const int row = rbase + i;
        float hc = tanhq(acc[i] + bhv);
        float z  = Zin[(size_t)row * H_DIM + n];
        float h  = Hf32[(size_t)row * H_DIM + n];
        float hn = z * h + (1.0f - z) * hc;
        Hf32[(size_t)row * H_DIM + n]    = hn;
        Hf16out[(size_t)row * H_PAD + n] = (_Float16)hn;
    }
}

// ---------------- FC + softmax --------------------------------------------------
__global__ void fc_softmax(const float* __restrict__ h, const float* __restrict__ Wfc,
                           const float* __restrict__ bfc, float* __restrict__ out) {
    const int b = blockIdx.x;
    const int c = threadIdx.x;      // block of 16, first 12 active
    __shared__ float logits[C_OUT];
    if (c < C_OUT) {
        float s = bfc[c];
        const float* hr = h + (size_t)b * H_DIM;
        const float* wr = Wfc + (size_t)c * H_DIM;
        for (int k = 0; k < H_DIM; ++k) s += hr[k] * wr[k];
        logits[c] = s;
    }
    __syncthreads();
    if (c < C_OUT) {
        float mx = logits[0];
        for (int i = 1; i < C_OUT; ++i) mx = fmaxf(mx, logits[i]);
        float sum = 0.0f;
        for (int i = 0; i < C_OUT; ++i) sum += __expf(logits[i] - mx);
        out[(size_t)b * C_OUT + c] = __expf(logits[c] - mx) / sum;
    }
}

// ---------------- host side -----------------------------------------------------
static inline int cdiv(int a, int b) { return (a + b - 1) / b; }

extern "C" void kernel_launch(void* const* d_in, const int* in_sizes, int n_in,
                              void* d_out, int out_size, void* d_ws, size_t ws_size,
                              hipStream_t stream) {
    (void)in_sizes; (void)n_in; (void)out_size; (void)ws_size;

    // inputs in setup_inputs() order
    const float* x   = (const float*)d_in[0];
    const float* Wh0 = (const float*)d_in[1];  const float* bh0 = (const float*)d_in[2];
    const float* Wz0 = (const float*)d_in[3];  const float* bz0 = (const float*)d_in[4];
    const float* Wr0 = (const float*)d_in[5];  const float* br0 = (const float*)d_in[6];
    const float* Uh0 = (const float*)d_in[7];  const float* Uz0 = (const float*)d_in[8];
    const float* Ur0 = (const float*)d_in[9];
    const float* Wh1 = (const float*)d_in[10]; const float* bh1 = (const float*)d_in[11];
    const float* Wz1 = (const float*)d_in[12]; const float* bz1 = (const float*)d_in[13];
    const float* Wr1 = (const float*)d_in[14]; const float* br1 = (const float*)d_in[15];
    const float* Uh1 = (const float*)d_in[16]; const float* Uz1 = (const float*)d_in[17];
    const float* Ur1 = (const float*)d_in[18];
    const float* Wfc = (const float*)d_in[19]; const float* bfc = (const float*)d_in[20];
    float* out = (float*)d_out;

    // workspace carve-up (256B aligned)
    char* ws = (char*)d_ws;
    size_t off = 0;
    auto carve = [&](size_t bytes) -> char* {
        char* p = ws + off;
        off = (off + bytes + 255) & ~(size_t)255;
        return p;
    };
    _Float16* x16   = (_Float16*)carve((size_t)T_STEPS * B_DIM * D_PAD * 2);
    _Float16* Wz0h  = (_Float16*)carve((size_t)H_DIM * D_PAD * 2);
    _Float16* Wr0h  = (_Float16*)carve((size_t)H_DIM * D_PAD * 2);
    _Float16* Wh0h  = (_Float16*)carve((size_t)H_DIM * D_PAD * 2);
    _Float16* Uz0h  = (_Float16*)carve((size_t)H_DIM * H_PAD * 2);
    _Float16* Ur0h  = (_Float16*)carve((size_t)H_DIM * H_PAD * 2);
    _Float16* Uh0h  = (_Float16*)carve((size_t)H_DIM * H_PAD * 2);
    _Float16* Wz1h  = (_Float16*)carve((size_t)H_DIM * H_PAD * 2);
    _Float16* Wr1h  = (_Float16*)carve((size_t)H_DIM * H_PAD * 2);
    _Float16* Wh1h  = (_Float16*)carve((size_t)H_DIM * H_PAD * 2);
    _Float16* Uz1h  = (_Float16*)carve((size_t)H_DIM * H_PAD * 2);
    _Float16* Ur1h  = (_Float16*)carve((size_t)H_DIM * H_PAD * 2);
    _Float16* Uh1h  = (_Float16*)carve((size_t)H_DIM * H_PAD * 2);
    float*    h0_32 = (float*)   carve((size_t)B_DIM * H_DIM * 4);
    _Float16* h0_16 = (_Float16*)carve((size_t)B_DIM * H_PAD * 2);
    float*    h1_32 = (float*)   carve((size_t)B_DIM * H_DIM * 4);
    _Float16* h1_16 = (_Float16*)carve((size_t)B_DIM * H_PAD * 2);
    float*    z0    = (float*)   carve((size_t)B_DIM * H_DIM * 4);
    float*    z1    = (float*)   carve((size_t)B_DIM * H_DIM * 4);
    _Float16* P0    = (_Float16*)carve((size_t)B_DIM * H_PAD * 2);
    _Float16* P1    = (_Float16*)carve((size_t)B_DIM * H_PAD * 2);

    const int TB = 256;
    // stage x and weights in padded f16 (fresh every call: ws may be poisoned)
    {
        int n = T_STEPS * B_DIM * D_PAD;
        cvt_pad_f16<<<cdiv(n, TB), TB, 0, stream>>>(x, x16, T_STEPS * B_DIM, D_IN, D_PAD);
    }
    {
        int nW0 = H_DIM * D_PAD, g = cdiv(nW0, TB);
        cvt_pad_f16<<<g, TB, 0, stream>>>(Wz0, Wz0h, H_DIM, D_IN, D_PAD);
        cvt_pad_f16<<<g, TB, 0, stream>>>(Wr0, Wr0h, H_DIM, D_IN, D_PAD);
        cvt_pad_f16<<<g, TB, 0, stream>>>(Wh0, Wh0h, H_DIM, D_IN, D_PAD);
    }
    {
        int nU = H_DIM * H_PAD, g = cdiv(nU, TB);
        cvt_pad_f16<<<g, TB, 0, stream>>>(Uz0, Uz0h, H_DIM, H_DIM, H_PAD);
        cvt_pad_f16<<<g, TB, 0, stream>>>(Ur0, Ur0h, H_DIM, H_DIM, H_PAD);
        cvt_pad_f16<<<g, TB, 0, stream>>>(Uh0, Uh0h, H_DIM, H_DIM, H_PAD);
        cvt_pad_f16<<<g, TB, 0, stream>>>(Wz1, Wz1h, H_DIM, H_DIM, H_PAD);
        cvt_pad_f16<<<g, TB, 0, stream>>>(Wr1, Wr1h, H_DIM, H_DIM, H_PAD);
        cvt_pad_f16<<<g, TB, 0, stream>>>(Wh1, Wh1h, H_DIM, H_DIM, H_PAD);
        cvt_pad_f16<<<g, TB, 0, stream>>>(Uz1, Uz1h, H_DIM, H_DIM, H_PAD);
        cvt_pad_f16<<<g, TB, 0, stream>>>(Ur1, Ur1h, H_DIM, H_DIM, H_PAD);
        cvt_pad_f16<<<g, TB, 0, stream>>>(Uh1, Uh1h, H_DIM, H_DIM, H_PAD);
    }
    {
        int nf = B_DIM * H_DIM, nh = B_DIM * H_PAD;
        zero_f32<<<cdiv(nf, TB), TB, 0, stream>>>(h0_32, nf);
        zero_f32<<<cdiv(nf, TB), TB, 0, stream>>>(h1_32, nf);
        zero_f16<<<cdiv(nh, TB), TB, 0, stream>>>(h0_16, nh);
        zero_f16<<<cdiv(nh, TB), TB, 0, stream>>>(h1_16, nh);
        zero_f16<<<cdiv(nh, TB), TB, 0, stream>>>(P0, nh);
        zero_f16<<<cdiv(nh, TB), TB, 0, stream>>>(P1, nh);
    }

    // interleaved recurrence: layer0 step t, then layer1 step t
    const int GEMM_BLOCKS = (B_DIM / 16) * (H_DIM / 16) / 8;  // 800 tiles / 8 waves
    for (int t = 0; t < T_STEPS; ++t) {
        const _Float16* xt = x16 + (size_t)t * B_DIM * D_PAD;
        gru_gates<<<GEMM_BLOCKS, 256, 0, stream>>>(
            xt, D_PAD, Wz0h, Wr0h, bz0, br0, h0_16, Uz0h, Ur0h, h0_32, z0, P0);
        gru_update<<<GEMM_BLOCKS, 256, 0, stream>>>(
            xt, D_PAD, Wh0h, bh0, P0, Uh0h, z0, h0_32, h0_16);
        gru_gates<<<GEMM_BLOCKS, 256, 0, stream>>>(
            h0_16, H_PAD, Wz1h, Wr1h, bz1, br1, h1_16, Uz1h, Ur1h, h1_32, z1, P1);
        gru_update<<<GEMM_BLOCKS, 256, 0, stream>>>(
            h0_16, H_PAD, Wh1h, bh1, P1, Uh1h, z1, h1_32, h1_16);
    }

    fc_softmax<<<B_DIM, 16, 0, stream>>>(h1_32, Wfc, bfc, out);
}
